// LowRankSNN_59399397704164
// MI455X (gfx1250) — compile-verified, matrix operand order checked by reading, not run on hardware
//
#include <hip/hip_runtime.h>
#include <hip/hip_bf16.h>
#include <math.h>

typedef __attribute__((ext_vector_type(16))) _Float16 v16h;
typedef __attribute__((ext_vector_type(8)))  _Float16 v8h;
typedef __attribute__((ext_vector_type(8)))  float    v8f;

// Problem constants (from setup_inputs)
#define N_TOT   5000
#define NE      4000
#define T_STEPS 1000
// Tiling: rows padded to 5008 (313 blocks of 16), K padded to 5024 (157 blocks of 32).
#define RBLK    313
#define KBLK    157
#define KE_BLK  125          // 4000/32 : k-blocks 0..124 are E columns
#define KPAD    (KBLK*32)    // 5024
#define TILE_HALVES 512      // 16x32 f16 tile
#define SCALE   1024.0f      // pre-scale so f16 values avoid denormal range
#define INV_SCALE (1.0f/1024.0f)
#define MV_WAVES 8           // waves per matvec workgroup

// ---------------------------------------------------------------------------
// Pack conn (fp32, row-major 5000x5000) into WMMA-A-layout f16 tiles,
// pre-multiplied by block gain gp and by SCALE.
// Tile(rb,kb): 32 lanes x 16 halves; element (lane,h) = row rb*16+(lane&15),
// col kb*32 + K where K = h + (h>=8?8:0) + (lane>=16?8:0)  (ISA 16-bit A layout).
// ---------------------------------------------------------------------------
__global__ void pack_conn_kernel(const float* __restrict__ conn,
                                 _Float16* __restrict__ pk) {
    int gid  = blockIdx.x * 256 + threadIdx.x;
    int tile = gid >> 5;
    int lane = gid & 31;
    if (tile >= RBLK * KBLK) return;
    int rb = tile / KBLK;
    int kb = tile % KBLK;
    int row = rb * 16 + (lane & 15);
    int laneoff = (lane >= 16) ? 8 : 0;
    _Float16* dst = pk + (size_t)tile * TILE_HALVES + lane * 16;
    #pragma unroll
    for (int h = 0; h < 16; ++h) {
        int K   = h + ((h >= 8) ? 8 : 0) + laneoff;
        int col = kb * 32 + K;
        float v = 0.0f;
        if (row < N_TOT && col < N_TOT) {
            float gp = (row < NE) ? ((col < NE) ? 0.004069f : 0.02672f)
                                  : ((col < NE) ? 0.003276f : 0.02138f);
            v = conn[(size_t)row * N_TOT + col] * gp * SCALE;
        }
        dst[h] = (_Float16)v;
    }
}

// ---------------------------------------------------------------------------
// Init: zero state, zero Out[], write t=0 output columns, zero f16 spikes.
// ---------------------------------------------------------------------------
__global__ void init_kernel(float* sA, float* sB, float* sG, float* phase,
                            _Float16* spkH,
                            float* outOut, float* outV, float* outG, float* outSpk) {
    int i = blockIdx.x * 256 + threadIdx.x;
    if (i < KPAD) spkH[i] = (_Float16)0.0f;
    if (i < T_STEPS) outOut[i] = 0.0f;
    if (i < N_TOT) {
        sA[i] = 0.0f; sB[i] = 0.0f; sG[i] = 0.0f; phase[i] = 0.0f;
        size_t o = (size_t)i * T_STEPS;
        outV[o]   = -58.5f;   // theta2V(0)
        outG[o]   = 0.0f;
        outSpk[o] = 0.0f;
    }
}

// ---------------------------------------------------------------------------
// Load one WMMA operand pair for k-block kb: A from packed tiles (two b128),
// B = spike vector broadcast into all 16 columns (two b128 from f16 vector).
// ---------------------------------------------------------------------------
__device__ __forceinline__ void load_ab(const _Float16* __restrict__ tilebase,
                                        const _Float16* __restrict__ spkH,
                                        int kb, int laneoff,
                                        v16h& a, v16h& b) {
    a = *(const v16h*)(tilebase + (size_t)kb * TILE_HALVES);
    int c = kb * 32 + laneoff;
    v8h blo = *(const v8h*)(spkH + c);        // K = c..c+7
    v8h bhi = *(const v8h*)(spkH + c + 16);   // K = c+16..c+23
    b = __builtin_shufflevector(blo, bhi,
        0,1,2,3,4,5,6,7,8,9,10,11,12,13,14,15);
}

// ---------------------------------------------------------------------------
// GEMV via WMMA f32_16x16x32_f16. One 16-row block per workgroup (8 waves);
// waves stride over k-blocks. Spike vector broadcast into all 16 B columns
// -> every D column holds the row sums. E-column (kb<125) and I-column
// partial sums kept in separate accumulators; the E chain is split into two
// independent accumulators for WMMA pipelining. All loop bounds are scalar
// (wave id forced into an SGPR) so no WMMA executes under a partial EXEC.
// ---------------------------------------------------------------------------
__global__ void __launch_bounds__(MV_WAVES * 32)
matvec_kernel(const _Float16* __restrict__ pk,
              const _Float16* __restrict__ spkH,
              float* __restrict__ mX, float* __restrict__ mY) {
    const int rb   = blockIdx.x;
    const int wave = __builtin_amdgcn_readfirstlane(threadIdx.x >> 5);  // SGPR
    const int lane = threadIdx.x & 31;
    const int laneoff = (lane >= 16) ? 8 : 0;
    const _Float16* tilebase = pk + (size_t)rb * KBLK * TILE_HALVES + lane * 16;

    v8f accE0 = {}, accE1 = {}, accI = {};

    // E columns (kb in [0,125)): two interleaved independent WMMA chains.
    int kb = wave;
    for (; kb + MV_WAVES < KE_BLK; kb += 2 * MV_WAVES) {
        v16h a0, b0, a1, b1;
        load_ab(tilebase, spkH, kb,            laneoff, a0, b0);
        load_ab(tilebase, spkH, kb + MV_WAVES, laneoff, a1, b1);
        __builtin_prefetch(tilebase + (size_t)(kb + 2 * MV_WAVES) * TILE_HALVES, 0, 3);
        accE0 = __builtin_amdgcn_wmma_f32_16x16x32_f16(
                    false, a0, false, b0, (short)0, accE0, false, false);
        accE1 = __builtin_amdgcn_wmma_f32_16x16x32_f16(
                    false, a1, false, b1, (short)0, accE1, false, false);
    }
    if (kb < KE_BLK) {
        v16h a0, b0;
        load_ab(tilebase, spkH, kb, laneoff, a0, b0);
        accE0 = __builtin_amdgcn_wmma_f32_16x16x32_f16(
                    false, a0, false, b0, (short)0, accE0, false, false);
    }

    // I columns (kb in [125,157)): ~4 blocks per wave.
    for (int kbi = KE_BLK + wave; kbi < KBLK; kbi += MV_WAVES) {
        v16h a0, b0;
        load_ab(tilebase, spkH, kbi, laneoff, a0, b0);
        accI = __builtin_amdgcn_wmma_f32_16x16x32_f16(
                   false, a0, false, b0, (short)0, accI, false, false);
    }

    v8f accE = accE0 + accE1;

    // D layout: lane L holds column L%16, rows v + (L>=16 ? 8 : 0).
    // All columns identical -> lanes 0 and 16 carry the full 16-row result.
    __shared__ float red[MV_WAVES][2][16];
    if ((lane & 15) == 0) {
        int rbase = (lane >= 16) ? 8 : 0;
        #pragma unroll
        for (int v = 0; v < 8; ++v) {
            red[wave][0][rbase + v] = accE[v];
            red[wave][1][rbase + v] = accI[v];
        }
    }
    __syncthreads();
    if (threadIdx.x < 32) {
        int r   = threadIdx.x & 15;
        int sel = threadIdx.x >> 4;
        float s = 0.0f;
        #pragma unroll
        for (int w = 0; w < MV_WAVES; ++w) s += red[w][sel][r];
        int row = rb * 16 + r;
        if (row < N_TOT) (sel ? mY : mX)[row] = s * INV_SCALE;
    }
}

// ---------------------------------------------------------------------------
// Per-neuron ODE + phase update + output writes + Out[t] reduction.
// mX = gp0*mEE (E rows) / gp2*mIE (I rows); mY = gp1*mEI / gp3*mII.
// ---------------------------------------------------------------------------
__global__ void __launch_bounds__(256)
update_kernel(const float* __restrict__ dtp,
              const float* __restrict__ Input,
              const float* __restrict__ W,
              const int* __restrict__ taudEp, const int* __restrict__ taudIp,
              float* sA, float* sB, float* sG, float* phase,
              const float* __restrict__ mX, const float* __restrict__ mY,
              _Float16* spkH,
              float* outOut, float* outV, float* outG, float* outSpk,
              int t) {
    int i = blockIdx.x * 256 + threadIdx.x;
    float contrib = 0.0f;
    if (i < N_TOT) {
        const float dt  = dtp[0];
        const bool isE  = (i < NE);
        const float itau = 1.0f / (isE ? (float)taudEp[0] : (float)taudIp[0]);
        const float gl   = isE ? 0.08f : 0.1f;
        const float C0 = 2.0f / 7.0f;        //  2/(V_T - V_R)
        const float C1 = 117.0f / 7.0f;      // (-V_R - V_T)/(V_T - V_R)
        const float C2 = -23.0f / 7.0f;      // (-140 - V_R - V_T)/(V_T - V_R)
        const float PI_F = 3.14159265358979f;

        float mx = mX[i], my = mY[i];
        float a = sA[i]; a += (-a * itau + mx) * dt;
        float b = sB[i]; b += (-b * itau + my) * dt;
        float g = sG[i]; g += (-g * itau + mx + my) * dt;

        float ph = phase[i];
        float cp = cosf(ph), sp = sinf(ph);
        float inp = Input[(size_t)i * T_STEPS + t];
        float ph2 = ph + (-gl * cp + C0 * (1.0f + cp) * inp
                          + a * (C1 * (1.0f + cp) - sp)
                          + b * (C2 * (1.0f + cp) - sp)) * dt;
        float spk = 0.0f;
        if (ph2 >= PI_F) { spk = 1.0f; ph2 -= 2.0f * PI_F; }

        sA[i] = a; sB[i] = b; sG[i] = g; phase[i] = ph2;
        spkH[i] = (_Float16)spk;

        size_t o = (size_t)i * T_STEPS + t;
        outG[o]   = g;
        outV[o]   = -58.5f + 3.5f * tanf(ph2 * 0.5f);
        outSpk[o] = spk;
        contrib = g * W[i] * (1.0f / (float)N_TOT);
    }
    __shared__ float sred[256];
    sred[threadIdx.x] = contrib;
    __syncthreads();
    for (int st = 128; st > 0; st >>= 1) {
        if (threadIdx.x < st) sred[threadIdx.x] += sred[threadIdx.x + st];
        __syncthreads();
    }
    if (threadIdx.x == 0) atomicAdd(&outOut[t], sred[0]);
}

// ---------------------------------------------------------------------------
extern "C" void kernel_launch(void* const* d_in, const int* in_sizes, int n_in,
                              void* d_out, int out_size, void* d_ws, size_t ws_size,
                              hipStream_t stream) {
    const float* dtp   = (const float*)d_in[0];
    const float* Input = (const float*)d_in[1];
    const float* conn  = (const float*)d_in[2];
    const float* W     = (const float*)d_in[3];
    const int*   taudE = (const int*)d_in[6];
    const int*   taudI = (const int*)d_in[7];

    // Output layout: Out(1000) | V(5M) | g(5M) | spk(5M)
    float* out    = (float*)d_out;
    float* outOut = out;
    float* outV   = out + T_STEPS;
    float* outG   = outV + (size_t)N_TOT * T_STEPS;
    float* outSpk = outG + (size_t)N_TOT * T_STEPS;

    // Workspace layout (packed conn first; 1024B-aligned tail).
    const size_t PK_HALVES = (size_t)RBLK * KBLK * TILE_HALVES;   // 25,160,192
    char* ws = (char*)d_ws;
    _Float16* pk = (_Float16*)ws;
    float* f     = (float*)(ws + PK_HALVES * sizeof(_Float16));
    float* sA    = f;            // 5000
    float* sB    = sA + N_TOT;   // 5000
    float* sG    = sB + N_TOT;   // 5000
    float* phase = sG + N_TOT;   // 5000
    float* mX    = phase + N_TOT;   // 5008
    float* mY    = mX + 5008;       // 5008
    _Float16* spkH = (_Float16*)(mY + 5008);   // 5024 halves, 16B-aligned

    // One-time (per launch) pack + init.
    {
        int threads = RBLK * KBLK * 32;
        pack_conn_kernel<<<(threads + 255) / 256, 256, 0, stream>>>(conn, pk);
        init_kernel<<<(KPAD + 255) / 256, 256, 0, stream>>>(
            sA, sB, sG, phase, spkH, outOut, outV, outG, outSpk);
    }

    // Sequential scan: 999 steps.
    for (int t = 1; t < T_STEPS; ++t) {
        matvec_kernel<<<RBLK, MV_WAVES * 32, 0, stream>>>(pk, spkH, mX, mY);
        update_kernel<<<(N_TOT + 255) / 256, 256, 0, stream>>>(
            dtp, Input, W, taudE, taudI,
            sA, sB, sG, phase, mX, mY, spkH,
            outOut, outV, outG, outSpk, t);
    }
}
